// MambaFusionModule_2010044695185
// MI455X (gfx1250) — matrix operand order repaired and checked by reading, compiled, not run on hardware
//
#include <hip/hip_runtime.h>
#include <hip/hip_bf16.h>
#include <math.h>

// ---------------- problem constants ----------------
#define BB      2
#define LL      2048
#define DMODEL  512
#define DINNER  1024
#define DSTATE  16
#define DTRANK  32
#define KCONV   4
#define MROWS   (BB * LL)          // 4096 token rows
#define TCHUNK  32                 // scan LDS staging depth

// ---------------- vector types ----------------
typedef __attribute__((ext_vector_type(16))) __bf16          v16bf;
typedef __attribute__((ext_vector_type(16))) unsigned short  v16u;
typedef __attribute__((ext_vector_type(8)))  unsigned short  v8u;
typedef __attribute__((ext_vector_type(8)))  float           v8f;

// ---------------- small device helpers ----------------
__device__ __forceinline__ float siluf(float x) { return x / (1.f + __expf(-x)); }
__device__ __forceinline__ float softplusf(float x) { return (x > 20.f) ? x : log1pf(__expf(x)); }
__device__ __forceinline__ unsigned short f2bf(float f) {
  unsigned int u = __float_as_uint(f);
  unsigned int r = (u + 0x7FFFu + ((u >> 16) & 1u)) >> 16;   // round-to-nearest-even
  return (unsigned short)r;
}

// =====================================================================
// Register-blocked BF16 WMMA GEMM:
//   C[M,N] = A[M,K] (row-major bf16) * W[N,K]^T
// One wave32 per (16*BM) x (16*BN) tile: BM*BN v8f accumulators,
// BM + BN fragments per 32-deep K chunk -> BM*BN wmma per (BM+BN)*2
// b128 loads.  Fragment layouts per CDNA5 ISA 7.12.2:
//   A: lane l -> row l&15; e0..7 = K kh..kh+7, e8..15 = K 16+kh..23+kh,
//      kh = 8*(l>=16)            (two contiguous 16B loads)
//   B: lane l -> col l&15; e0..7 = K kh..kh+7, e8..15 = K 16+kh..23+kh
//      (B[k][n] = W[n][k] -> two contiguous 16B loads from W row)
//   C/D: VGPR v -> M = v + 8*(l>=16), N = l&15
// Epilogue MODE: 0 plain f32; 1 in_proj split (u | silu(z));
//                2 xdbl f32 + bf16 copy of first 32 cols (dt);
//                3 softplus(val + bias[n]) -> delta
// =====================================================================
template <int MODE, int BM, int BN>
__global__ __launch_bounds__(32)
void gemm_wmma_bf16(const unsigned short* __restrict__ A,
                    const unsigned short* __restrict__ W,
                    int M, int N, int K,
                    float* __restrict__ out0,
                    float* __restrict__ out1,
                    unsigned short* __restrict__ outbf,
                    const float* __restrict__ bias) {
  const int lane = threadIdx.x;
  const int tm   = blockIdx.y * (16 * BM);
  const int tn   = blockIdx.x * (16 * BN);
  const int half = lane >> 4;      // 0 | 1
  const int r    = lane & 15;
  const int kh   = half * 8;

  v8f acc[BM][BN] = {};

  for (int k0 = 0; k0 < K; k0 += 32) {
    v16u af[BM], bf[BN];
#pragma unroll
    for (int i = 0; i < BM; ++i) {
      const unsigned short* ap = A + (size_t)(tm + i * 16 + r) * K + (k0 + kh);
      v8u lo = *reinterpret_cast<const v8u*>(ap);
      v8u hi = *reinterpret_cast<const v8u*>(ap + 16);
#pragma unroll
      for (int e = 0; e < 8; ++e) { af[i][e] = lo[e]; af[i][e + 8] = hi[e]; }
    }
#pragma unroll
    for (int j = 0; j < BN; ++j) {
      const unsigned short* bp = W + (size_t)(tn + j * 16 + r) * K + (k0 + kh);
      v8u lo = *reinterpret_cast<const v8u*>(bp);
      v8u hi = *reinterpret_cast<const v8u*>(bp + 16);
#pragma unroll
      for (int e = 0; e < 8; ++e) { bf[j][e] = lo[e]; bf[j][e + 8] = hi[e]; }
    }
#pragma unroll
    for (int i = 0; i < BM; ++i)
#pragma unroll
      for (int j = 0; j < BN; ++j)
        acc[i][j] = __builtin_amdgcn_wmma_f32_16x16x32_bf16(
            false, __builtin_bit_cast(v16bf, af[i]),
            false, __builtin_bit_cast(v16bf, bf[j]),
            (short)0, acc[i][j], false, false);
  }

#pragma unroll
  for (int i = 0; i < BM; ++i) {
#pragma unroll
    for (int j = 0; j < BN; ++j) {
#pragma unroll
      for (int v = 0; v < 8; ++v) {
        const int m = tm + i * 16 + v + half * 8;
        const int n = tn + j * 16 + r;
        float val = acc[i][j][v];
        if (MODE == 0) {
          out0[(size_t)m * N + n] = val;
        } else if (MODE == 1) {            // in_proj: [u | z] -> u, silu(z)=g
          if (n < DINNER) out0[(size_t)m * DINNER + n] = val;
          else            out1[(size_t)m * DINNER + (n - DINNER)] = siluf(val);
        } else if (MODE == 2) {            // xdbl; dt slice also as bf16
          out0[(size_t)m * N + n] = val;
          if (n < DTRANK) outbf[(size_t)m * DTRANK + n] = f2bf(val);
        } else if (MODE == 3) {            // delta = softplus(dtproj + dt_b)
          out0[(size_t)m * N + n] = softplusf(val + bias[n]);
        }
      }
    }
  }
}

// =====================================================================
// LayerNorm over rows of width 512. res != nullptr -> normalize (x+res).
// =====================================================================
__global__ __launch_bounds__(256)
void ln_kernel(const float* __restrict__ x, const float* __restrict__ res,
               const float* __restrict__ w, const float* __restrict__ b,
               unsigned short* __restrict__ out_bf, float* __restrict__ out_f) {
  __shared__ float sm[256];
  const int row = blockIdx.x;
  const int t   = threadIdx.x;
  const size_t base = (size_t)row * DMODEL;
  float a0 = x[base + t], a1 = x[base + t + 256];
  if (res) { a0 += res[base + t]; a1 += res[base + t + 256]; }
  sm[t] = a0 + a1;
  __syncthreads();
  for (int s = 128; s > 0; s >>= 1) { if (t < s) sm[t] += sm[t + s]; __syncthreads(); }
  const float mean = sm[0] * (1.f / DMODEL);
  __syncthreads();
  const float d0 = a0 - mean, d1 = a1 - mean;
  sm[t] = d0 * d0 + d1 * d1;
  __syncthreads();
  for (int s = 128; s > 0; s >>= 1) { if (t < s) sm[t] += sm[t + s]; __syncthreads(); }
  const float rstd = rsqrtf(sm[0] * (1.f / DMODEL) + 1e-5f);
  const float o0 = d0 * rstd * w[t] + b[t];
  const float o1 = d1 * rstd * w[t + 256] + b[t + 256];
  if (out_bf) { out_bf[base + t] = f2bf(o0); out_bf[base + t + 256] = f2bf(o1); }
  if (out_f)  { out_f[base + t] = o0;        out_f[base + t + 256] = o1; }
}

// =====================================================================
// Depthwise causal conv (K=4) + SiLU.  rev=1 reads u time-reversed.
// =====================================================================
__global__ __launch_bounds__(256)
void conv_silu_kernel(const float* __restrict__ u, const float* __restrict__ w,
                      const float* __restrict__ bias,
                      float* __restrict__ uc, unsigned short* __restrict__ ucbf,
                      int rev) {
  const size_t idx = (size_t)blockIdx.x * blockDim.x + threadIdx.x;
  if (idx >= (size_t)BB * LL * DINNER) return;
  const int d = (int)(idx % DINNER);
  const int t = (int)((idx / DINNER) % LL);
  const int b = (int)(idx / ((size_t)DINNER * LL));
  float acc = bias[d];
#pragma unroll
  for (int j = 0; j < KCONV; ++j) {
    const int tt = t - (KCONV - 1) + j;
    if (tt >= 0) {
      const int s = rev ? (LL - 1 - tt) : tt;
      acc += w[d * KCONV + j] * u[((size_t)b * LL + s) * DINNER + d];
    }
  }
  const float v = siluf(acc);
  uc[idx] = v;
  ucbf[idx] = f2bf(v);
}

// =====================================================================
// Selective scan with LDS staging.  Block = 256 threads = 16 channels
// (d) x 16 state lanes (n) for one batch b.  TCHUNK timesteps of
// delta/uc (16 ch) and B/C (16+16) are staged into LDS with coalesced
// loads; the recurrence then runs from LDS:
//   h = exp(delta*A)*h + delta*u*B ;  y = sum_n h*C  (shfl_xor tree)
// Forward pass stages y+u*D; backward pass combines, gates with g,
// emits bf16 rows for the out_proj WMMA GEMM.
// =====================================================================
__global__ __launch_bounds__(256)
void scan_kernel(const float* __restrict__ uc, const float* __restrict__ delta,
                 const float* __restrict__ xdbl,
                 const float* __restrict__ A_log, const float* __restrict__ D,
                 const float* __restrict__ g,
                 float* __restrict__ ystage, unsigned short* __restrict__ ybf,
                 int rev) {
  __shared__ float sdelta[TCHUNK * 16];
  __shared__ float suc[TCHUNK * 16];
  __shared__ float sbc[TCHUNK * 32];

  const int b    = blockIdx.x >> 6;           // DINNER/16 = 64 blocks per batch
  const int dblk = blockIdx.x & 63;
  const int tid  = threadIdx.x;
  const int n    = tid & 15;
  const int dl   = tid >> 4;
  const int d0   = dblk * 16;
  const int d    = d0 + dl;

  const float Av = -__expf(A_log[d * DSTATE + n]);
  const float Dv = D[d];
  float h = 0.f;

  for (int t0 = 0; t0 < LL; t0 += TCHUNK) {
    // ---- stage TCHUNK steps into LDS (coalesced) ----
#pragma unroll
    for (int q = 0; q < (TCHUNK * 16) / 256; ++q) {
      const int k = tid + q * 256;
      const int i = k >> 4, j = k & 15;
      const size_t src = ((size_t)b * LL + t0 + i) * DINNER + d0 + j;
      sdelta[k] = delta[src];
      suc[k]    = uc[src];
    }
#pragma unroll
    for (int q = 0; q < (TCHUNK * 32) / 256; ++q) {
      const int k = tid + q * 256;
      const int i = k >> 5, c = k & 31;
      sbc[k] = xdbl[((size_t)b * LL + t0 + i) * 64 + DTRANK + c];
    }
    __syncthreads();

    // ---- recurrence over the chunk ----
    for (int s = 0; s < TCHUNK; ++s) {
      const float dt = sdelta[s * 16 + dl];
      const float ut = suc[s * 16 + dl];
      const float Bt = sbc[s * 32 + n];
      const float Ct = sbc[s * 32 + 16 + n];
      h = __expf(dt * Av) * h + (dt * ut) * Bt;
      float y = h * Ct;
      y += __shfl_xor(y, 1, 32);
      y += __shfl_xor(y, 2, 32);
      y += __shfl_xor(y, 4, 32);
      y += __shfl_xor(y, 8, 32);
      if (n == 0) {
        const int t = t0 + s;
        const float val = y + ut * Dv;
        if (!rev) {
          ystage[((size_t)b * LL + t) * DINNER + d] = val;
        } else {
          const size_t orow = (size_t)b * LL + (LL - 1 - t);
          const float tot = (ystage[orow * DINNER + d] + val) * g[orow * DINNER + d];
          ybf[orow * DINNER + d] = f2bf(tot);
        }
      }
    }
    __syncthreads();
  }
}

// f32 -> bf16 bulk convert (weights)
__global__ __launch_bounds__(256)
void to_bf16_kernel(const float* __restrict__ src, unsigned short* __restrict__ dst, int n) {
  const int i = blockIdx.x * blockDim.x + threadIdx.x;
  if (i < n) dst[i] = f2bf(src[i]);
}

// =====================================================================
extern "C" void kernel_launch(void* const* d_in, const int* in_sizes, int n_in,
                              void* d_out, int out_size, void* d_ws, size_t ws_size,
                              hipStream_t stream) {
  const float* x        = (const float*)d_in[0];
  const float* ln1_w    = (const float*)d_in[1];
  const float* ln1_b    = (const float*)d_in[2];
  const float* ln2_w    = (const float*)d_in[3];
  const float* ln2_b    = (const float*)d_in[4];
  const float* in_proj  = (const float*)d_in[5];
  const float* convw[2]  = {(const float*)d_in[6],  (const float*)d_in[13]};
  const float* convb[2]  = {(const float*)d_in[7],  (const float*)d_in[14]};
  const float* xprojw[2] = {(const float*)d_in[8],  (const float*)d_in[15]};
  const float* dtw[2]    = {(const float*)d_in[9],  (const float*)d_in[16]};
  const float* dtb[2]    = {(const float*)d_in[10], (const float*)d_in[17]};
  const float* Alog[2]   = {(const float*)d_in[11], (const float*)d_in[18]};
  const float* Dp[2]     = {(const float*)d_in[12], (const float*)d_in[19]};
  const float* out_proj = (const float*)d_in[20];
  float* out = (float*)d_out;

  // ---- bump allocator over workspace ----
  char* cur = (char*)d_ws;
  auto alloc = [&](size_t bytes) -> void* {
    void* p = (void*)cur;
    cur += (bytes + 255) & ~(size_t)255;
    return p;
  };
  unsigned short* xn_bf   = (unsigned short*)alloc((size_t)MROWS * DMODEL * 2);
  unsigned short* w_in_bf = (unsigned short*)alloc((size_t)2 * DINNER * DMODEL * 2);
  unsigned short* w_xp_bf[2] = {(unsigned short*)alloc((size_t)64 * DINNER * 2),
                                (unsigned short*)alloc((size_t)64 * DINNER * 2)};
  unsigned short* w_dt_bf[2] = {(unsigned short*)alloc((size_t)DINNER * DTRANK * 2),
                                (unsigned short*)alloc((size_t)DINNER * DTRANK * 2)};
  unsigned short* w_out_bf = (unsigned short*)alloc((size_t)DMODEL * DINNER * 2);
  float* u_buf  = (float*)alloc((size_t)MROWS * DINNER * 4);
  float* g_buf  = (float*)alloc((size_t)MROWS * DINNER * 4);
  float* uc[2]  = {(float*)alloc((size_t)MROWS * DINNER * 4),
                   (float*)alloc((size_t)MROWS * DINNER * 4)};
  unsigned short* ucbf[2] = {(unsigned short*)alloc((size_t)MROWS * DINNER * 2),
                             (unsigned short*)alloc((size_t)MROWS * DINNER * 2)};
  float* xdbl[2] = {(float*)alloc((size_t)MROWS * 64 * 4),
                    (float*)alloc((size_t)MROWS * 64 * 4)};
  unsigned short* dtbf[2] = {(unsigned short*)alloc((size_t)MROWS * DTRANK * 2),
                             (unsigned short*)alloc((size_t)MROWS * DTRANK * 2)};
  float* delta[2] = {(float*)alloc((size_t)MROWS * DINNER * 4),
                     (float*)alloc((size_t)MROWS * DINNER * 4)};
  // aliases (lifetimes disjoint): u dead after convs; ucbf[0] dead after
  // x_proj fw GEMM; uc[0] dead once both scans have consumed it.
  float* ystage          = u_buf;                       // fw scan staging
  unsigned short* ybf    = ucbf[0];                     // out_proj A operand
  float* out_pre         = uc[0];                       // pre-LN2 output

  // ---- 1. weights -> bf16 ----
  {
    struct { const float* s; unsigned short* d; int n; } cv[6] = {
      {in_proj,   w_in_bf,    2 * DINNER * DMODEL},
      {xprojw[0], w_xp_bf[0], 64 * DINNER},
      {xprojw[1], w_xp_bf[1], 64 * DINNER},
      {dtw[0],    w_dt_bf[0], DINNER * DTRANK},
      {dtw[1],    w_dt_bf[1], DINNER * DTRANK},
      {out_proj,  w_out_bf,   DMODEL * DINNER},
    };
    for (int i = 0; i < 6; ++i)
      to_bf16_kernel<<<(cv[i].n + 255) / 256, 256, 0, stream>>>(cv[i].s, cv[i].d, cv[i].n);
  }

  // ---- 2. LN1 -> xn (bf16) ----
  ln_kernel<<<MROWS, 256, 0, stream>>>(x, nullptr, ln1_w, ln1_b, xn_bf, nullptr);

  // ---- 3. in_proj GEMM, 64x64/wave (split epilogue: u | silu(z)=g) ----
  gemm_wmma_bf16<1, 4, 4><<<dim3((2 * DINNER) / 64, MROWS / 64), 32, 0, stream>>>(
      xn_bf, w_in_bf, MROWS, 2 * DINNER, DMODEL, u_buf, g_buf, nullptr, nullptr);

  // ---- 4. causal conv + SiLU, both directions ----
  {
    const size_t tot = (size_t)BB * LL * DINNER;
    const int blocks = (int)((tot + 255) / 256);
    conv_silu_kernel<<<blocks, 256, 0, stream>>>(u_buf, convw[0], convb[0], uc[0], ucbf[0], 0);
    conv_silu_kernel<<<blocks, 256, 0, stream>>>(u_buf, convw[1], convb[1], uc[1], ucbf[1], 1);
  }

  // ---- 5. per-branch x_proj (16x16, keep occupancy: N=64) and dt GEMMs ----
  for (int br = 0; br < 2; ++br) {
    gemm_wmma_bf16<2, 1, 1><<<dim3(64 / 16, MROWS / 16), 32, 0, stream>>>(
        ucbf[br], w_xp_bf[br], MROWS, 64, DINNER, xdbl[br], nullptr, dtbf[br], nullptr);
    gemm_wmma_bf16<3, 2, 2><<<dim3(DINNER / 32, MROWS / 32), 32, 0, stream>>>(
        dtbf[br], w_dt_bf[br], MROWS, DINNER, DTRANK, delta[br], nullptr, nullptr, dtb[br]);
  }

  // ---- 6. selective scans (fw stages, bw combines + gates + bf16) ----
  scan_kernel<<<BB * (DINNER / 16), 256, 0, stream>>>(
      uc[0], delta[0], xdbl[0], Alog[0], Dp[0], g_buf, ystage, nullptr, 0);
  scan_kernel<<<BB * (DINNER / 16), 256, 0, stream>>>(
      uc[1], delta[1], xdbl[1], Alog[1], Dp[1], g_buf, ystage, ybf, 1);

  // ---- 7. out_proj GEMM, 32x32/wave ----
  gemm_wmma_bf16<0, 2, 2><<<dim3(DMODEL / 32, MROWS / 32), 32, 0, stream>>>(
      ybf, w_out_bf, MROWS, DMODEL, DINNER, out_pre, nullptr, nullptr, nullptr);

  // ---- 8. residual + LN2 -> d_out ----
  ln_kernel<<<MROWS, 256, 0, stream>>>(out_pre, x, ln2_w, ln2_b, nullptr, out);

  (void)in_sizes; (void)n_in; (void)out_size; (void)ws_size;
}
